// RouterBase_6622839571060
// MI455X (gfx1250) — matrix operand order repaired and checked by reading, not compiled, not run on hardware
//
#include <hip/hip_runtime.h>
#include <hip/hip_bf16.h>
#include <stdint.h>

typedef __attribute__((ext_vector_type(16))) __bf16        v16bf;
typedef __attribute__((ext_vector_type(8)))  unsigned int  v8u;
typedef __attribute__((ext_vector_type(8)))  float         v8f;

constexpr int T = 16384;   // tokens
constexpr int H = 4096;    // hidden
constexpr int E = 64;      // experts
constexpr int K = 8;       // topk

// d_out layout (floats, reference tuple order)
constexpr size_t OFF_LOGITS = 0;
constexpr size_t OFF_PROBS  = (size_t)T * E;              // 1048576
constexpr size_t OFF_TIDX   = OFF_PROBS + (size_t)T * K;  // 1179648
constexpr size_t OFF_GROUP  = OFF_TIDX + (size_t)T * K;   // 1310720
constexpr size_t OFF_SORT   = OFF_GROUP + E;              // 1310784
constexpr size_t OFF_TPE    = OFF_SORT + (size_t)T * K;   // 1441856
constexpr size_t OFF_SCAL   = OFF_TPE + E;                // 1441920 (lb, zl)

// ---------------- fp32 -> bf16 helpers ----------------
// Exact RNE (used once, in the W pre-conversion kernel)
__device__ __forceinline__ unsigned short f2bf_rne(float f) {
  unsigned int u = __builtin_bit_cast(unsigned int, f);
  unsigned int r = u + 0x7FFFu + ((u >> 16) & 1u);
  return (unsigned short)(r >> 16);
}

// Fast path for the GEMM inner loop: round-to-nearest + v_perm_b32 packs two
// bf16 into one dword: lo -> bits[15:0], hi -> bits[31:16].
__device__ __forceinline__ unsigned int pack2_bf16(float lo, float hi) {
  unsigned int ulo = __builtin_bit_cast(unsigned int, lo) + 0x8000u;
  unsigned int uhi = __builtin_bit_cast(unsigned int, hi) + 0x8000u;
  return __builtin_amdgcn_perm(uhi, ulo, 0x07060302u);
}

// ---------------- Kernel 0: W (fp32) -> Wbf (bf16 bits) ----------------
__global__ __launch_bounds__(256) void convert_w(const float* __restrict__ W,
                                                 unsigned short* __restrict__ Wbf) {
  int i = (blockIdx.x * 256 + threadIdx.x) * 4;   // element index, covers E*H
  float4 v = *(const float4*)(W + i);
  unsigned int d0 = (unsigned int)f2bf_rne(v.x) | ((unsigned int)f2bf_rne(v.y) << 16);
  unsigned int d1 = (unsigned int)f2bf_rne(v.z) | ((unsigned int)f2bf_rne(v.w) << 16);
  uint2 o; o.x = d0; o.y = d1;
  *(uint2*)(Wbf + i) = o;
}

// ---------------- Kernel 1: logits = x @ W^T via bf16 WMMA ----------------
// grid 128, block 256 (8 waves). Wave w: tokens [t0, t0+16), all 64 experts.
// Software pipelined with an explicit scheduling fence so the compiler cannot
// sink the loads onto their consumers: all VMEM for the iteration issues up
// front (B first, HBM A-prefetch last), then convert + 4 WMMAs overlap them.
__global__ __launch_bounds__(256) void router_gemm(const float* __restrict__ x,
                                                   const unsigned short* __restrict__ Wbf,
                                                   float* __restrict__ logits) {
  const int lane = threadIdx.x & 31;
  const int wave = threadIdx.x >> 5;
  const int half = lane >> 4;     // lane group 0/1
  const int mn   = lane & 15;     // row (A) / col (B,C)
  const int t0   = (blockIdx.x * 8 + wave) * 16;

  const float* xrow = x + (size_t)(t0 + mn) * H;
  const unsigned short* wrow = Wbf + (size_t)mn * H;   // nt*16*H folds to const offset

  v8f acc[4] = {};                // 4 N-tiles of 16 experts

  // prologue: A data for it = 0
  // A fragment mapping: elems 0..7  -> K = kb + half*16 + 0..7
  //                     elems 8..15 -> K = kb + half*16 + 8 + 0..7
  const float4* a0 = (const float4*)(xrow + half * 16);
  const float4* a1 = (const float4*)(xrow + 8 + half * 16);
  float4 p = a0[0], q = a0[1], r = a1[0], s = a1[1];

  for (int it = 0; it < H / 32; ++it) {
    const int kb = it * 32;

    // --- VMEM issue group: B fragments first (fast, L2/WGP$-hot), then the
    //     next iteration's A (HBM stream, wraps on last iter -- harmless).
    v8u b0 = *(const v8u*)(wrow + 0 * 16 * H + kb + half * 16);
    v8u b1 = *(const v8u*)(wrow + 1 * 16 * H + kb + half * 16);
    v8u b2 = *(const v8u*)(wrow + 2 * 16 * H + kb + half * 16);
    v8u b3 = *(const v8u*)(wrow + 3 * 16 * H + kb + half * 16);

    const int kp = ((it + 1) * 32) & (H - 1);
    const float4* n0 = (const float4*)(xrow + kp + half * 16);
    const float4* n1 = (const float4*)(xrow + kp + 8 + half * 16);
    float4 pn = n0[0], qn = n0[1], rn = n1[0], sn = n1[1];

    __builtin_amdgcn_sched_barrier(0);   // loads stay issued above this point

    // convert current A (already resident) while loads are in flight
    v8u au;
    au[0] = pack2_bf16(p.x, p.y); au[1] = pack2_bf16(p.z, p.w);
    au[2] = pack2_bf16(q.x, q.y); au[3] = pack2_bf16(q.z, q.w);
    au[4] = pack2_bf16(r.x, r.y); au[5] = pack2_bf16(r.z, r.w);
    au[6] = pack2_bf16(s.x, s.y); au[7] = pack2_bf16(s.z, s.w);
    v16bf afrag = __builtin_bit_cast(v16bf, au);

    // matrix ops; in-order load completion => staged partial loadcnt waits
    acc[0] = __builtin_amdgcn_wmma_f32_16x16x32_bf16(
        false, afrag, false, __builtin_bit_cast(v16bf, b0), (short)0, acc[0], false, false);
    acc[1] = __builtin_amdgcn_wmma_f32_16x16x32_bf16(
        false, afrag, false, __builtin_bit_cast(v16bf, b1), (short)0, acc[1], false, false);
    acc[2] = __builtin_amdgcn_wmma_f32_16x16x32_bf16(
        false, afrag, false, __builtin_bit_cast(v16bf, b2), (short)0, acc[2], false, false);
    acc[3] = __builtin_amdgcn_wmma_f32_16x16x32_bf16(
        false, afrag, false, __builtin_bit_cast(v16bf, b3), (short)0, acc[3], false, false);

    p = pn; q = qn; r = rn; s = sn;
  }

  // C/D: VGPR rr -> M = rr + half*8 ; N = mn
#pragma unroll
  for (int nt = 0; nt < 4; ++nt) {
#pragma unroll
    for (int rr = 0; rr < 8; ++rr) {
      int row = t0 + rr + half * 8;
      logits[(size_t)row * E + nt * 16 + mn] = acc[nt][rr];
    }
  }
}

// ---------------- Kernel 2: per-token top-8, softmaxes, partials ----------------
// grid 128, block 128 (1 token / thread).
__global__ __launch_bounds__(128) void router_topk(const float* __restrict__ logits,
                                                   float* __restrict__ outProbs,
                                                   float* __restrict__ outIdx,
                                                   int*   __restrict__ flat,
                                                   int*   __restrict__ blockCounts,
                                                   float* __restrict__ probPartial,
                                                   float* __restrict__ lse2Partial) {
  __shared__ float srow[128][65];
  __shared__ float lsq[128];
  __shared__ int   hist[64];

  const int tid = threadIdx.x;
  const int t   = blockIdx.x * 128 + tid;

  if (tid < 64) hist[tid] = 0;
  __syncthreads();

  // load raw logits for this token
  const float4* lp = (const float4*)(logits + (size_t)t * E);
#pragma unroll
  for (int i = 0; i < 16; ++i) {
    float4 v = lp[i];
    srow[tid][i * 4 + 0] = v.x;
    srow[tid][i * 4 + 1] = v.y;
    srow[tid][i * 4 + 2] = v.z;
    srow[tid][i * 4 + 3] = v.w;
  }

  // top-8 on clipped scores; strict '>' => lowest index wins ties (clip ties too)
  unsigned long long used = 0ull;
  float tsc[8]; int tix[8];
#pragma unroll
  for (int k = 0; k < 8; ++k) {
    float best = -1e30f; int bi = 0;
    for (int e = 0; e < 64; ++e) {
      if (used & (1ull << e)) continue;
      float s = fminf(fmaxf(srow[tid][e], -2.f), 2.f);
      if (s > best) { best = s; bi = e; }
    }
    used |= (1ull << bi);
    tsc[k] = best; tix[k] = bi;
  }

  // softmax over the 8 selected (clipped) scores
  float m8 = tsc[0];
#pragma unroll
  for (int k = 1; k < 8; ++k) m8 = fmaxf(m8, tsc[k]);
  float ex[8], s8 = 0.f;
#pragma unroll
  for (int k = 0; k < 8; ++k) { ex[k] = __expf(tsc[k] - m8); s8 += ex[k]; }
  float inv8 = 1.f / s8;
#pragma unroll
  for (int k = 0; k < 8; ++k) {
    outProbs[(size_t)t * K + k] = ex[k] * inv8;
    outIdx[(size_t)t * K + k]   = (float)tix[k];
    flat[(size_t)t * K + k]     = tix[k];
    atomicAdd(&hist[tix[k]], 1);
  }

  // full softmax over raw logits + logsumexp^2 ; overwrite LDS row with probs
  float m = -1e30f;
  for (int e = 0; e < 64; ++e) m = fmaxf(m, srow[tid][e]);
  float s = 0.f;
  for (int e = 0; e < 64; ++e) s += __expf(srow[tid][e] - m);
  float lse = m + __logf(s);
  lsq[tid] = lse * lse;
  float inv = 1.f / s;
  for (int e = 0; e < 64; ++e) srow[tid][e] = __expf(srow[tid][e] - m) * inv;

  __syncthreads();

  // deterministic fixed-order reductions
  if (tid < 64) {
    float cs = 0.f;
    for (int r = 0; r < 128; ++r) cs += srow[r][tid];
    probPartial[(size_t)blockIdx.x * 64 + tid] = cs;
    blockCounts[(size_t)blockIdx.x * 64 + tid] = hist[tid];
  }
  if (tid == 0) {
    float a = 0.f;
    for (int r = 0; r < 128; ++r) a += lsq[r];
    lse2Partial[blockIdx.x] = a;
  }
}

// ---------------- Kernel 3: counts, cumsums, offsets, losses ----------------
__global__ __launch_bounds__(64) void router_finalize(const int* __restrict__ blockCounts,
                                                      const float* __restrict__ probPartial,
                                                      const float* __restrict__ lse2Partial,
                                                      int*   __restrict__ start,
                                                      float* __restrict__ outGroup,
                                                      float* __restrict__ outTpe,
                                                      float* __restrict__ outScalars) {
  __shared__ int   tpeL[64];
  __shared__ float spL[64];
  __shared__ int   baseL[64];
  const int e = threadIdx.x;

  int run = 0;
  for (int b = 0; b < 128; ++b) {
    start[b * 64 + e] = run;          // per-block exclusive prefix within expert
    run += blockCounts[b * 64 + e];
  }
  tpeL[e]   = run;
  outTpe[e] = (float)run;

  float sp = 0.f;
  for (int b = 0; b < 128; ++b) sp += probPartial[b * 64 + e];
  spL[e] = sp;
  __syncthreads();

  if (e == 0) {
    int base = 0;
    for (int i = 0; i < 64; ++i) {
      baseL[i] = base;
      base += tpeL[i];
      outGroup[i] = (float)base;      // inclusive cumsum
    }
    float lb = 0.f;
    for (int i = 0; i < 64; ++i)
      lb += ((float)tpeL[i] / (float)(T * K)) * (spL[i] / (float)T);
    outScalars[0] = lb * (float)E;
    float z = 0.f;
    for (int b = 0; b < 128; ++b) z += lse2Partial[b];
    outScalars[1] = z / (float)T;
  }
  __syncthreads();

  int bs = baseL[e];
  for (int b = 0; b < 128; ++b) start[b * 64 + e] += bs;
}

// ---------------- Kernel 4: stable counting-sort scatter (argsort) ----------------
// grid 128 (1024 flat slots each), block 64 (1 thread / expert).
__global__ __launch_bounds__(64) void router_scatter(const int* __restrict__ flat,
                                                     const int* __restrict__ start,
                                                     float* __restrict__ outIndices) {
  __shared__ int ids[1024];
  const int b = blockIdx.x;
  for (int i = threadIdx.x; i < 1024; i += 64) ids[i] = flat[(size_t)b * 1024 + i];
  __syncthreads();
  const int e = threadIdx.x;
  int pos = start[b * 64 + e];
  for (int i = 0; i < 1024; ++i)
    if (ids[i] == e) outIndices[pos++] = (float)(b * 1024 + i);
}

// ---------------- launcher ----------------
extern "C" void kernel_launch(void* const* d_in, const int* in_sizes, int n_in,
                              void* d_out, int out_size, void* d_ws, size_t ws_size,
                              hipStream_t stream) {
  const float* x = (const float*)d_in[0];
  const float* W = (const float*)d_in[1];
  float* out = (float*)d_out;

  // workspace carve-up
  unsigned short* Wbf = (unsigned short*)d_ws;           // E*H bf16 (512 KB)
  int*   flat        = (int*)(Wbf + (size_t)E * H);      // T*K ints
  int*   blockCounts = flat + (size_t)T * K;             // 128*64 ints
  int*   start       = blockCounts + 128 * 64;           // 128*64 ints
  float* probPartial = (float*)(start + 128 * 64);       // 128*64 floats
  float* lse2Partial = probPartial + 128 * 64;           // 128 floats

  convert_w<<<(E * H) / (256 * 4), 256, 0, stream>>>(W, Wbf);

  router_gemm<<<128, 256, 0, stream>>>(x, Wbf, out + OFF_LOGITS);

  router_topk<<<128, 128, 0, stream>>>(out + OFF_LOGITS,
                                       out + OFF_PROBS,
                                       out + OFF_TIDX,
                                       flat, blockCounts, probPartial, lse2Partial);

  router_finalize<<<1, 64, 0, stream>>>(blockCounts, probPartial, lse2Partial,
                                        start,
                                        out + OFF_GROUP,
                                        out + OFF_TPE,
                                        out + OFF_SCAL);

  router_scatter<<<128, 64, 0, stream>>>(flat, start, out + OFF_SORT);
}
